// PointCrossAttention_16106127360665
// MI455X (gfx1250) — matrix operand order repaired and checked by reading, not compiled, not run on hardware
//
#include <hip/hip_runtime.h>
#include <hip/hip_bf16.h>
#include <math.h>

// ---------------------------------------------------------------------------
// Types
// ---------------------------------------------------------------------------
typedef __bf16 bhalf;
typedef __attribute__((ext_vector_type(4)))  bhalf v4bf;
typedef __attribute__((ext_vector_type(8)))  bhalf v8bf;
typedef __attribute__((ext_vector_type(16))) bhalf v16bf;
typedef __attribute__((ext_vector_type(8)))  float v8f;

// Problem constants
#define BB   4
#define DD   128
#define HH   128
#define WW   128
#define FF   128
#define MM   4
#define DHD  32
#define PP   8
#define BQ   (BB * HH * WW)          // 65536
#define HID  (4 * DD)                // 512

// ---------------------------------------------------------------------------
// f32 -> bf16 conversion (weights)
// ---------------------------------------------------------------------------
__global__ __launch_bounds__(256) void cvt_f32_bf16(const float* __restrict__ s,
                                                    bhalf* __restrict__ d, long n) {
  long i = (long)blockIdx.x * 256 + threadIdx.x;
  if (i < n) d[i] = (bhalf)s[i];
}

// ---------------------------------------------------------------------------
// bev_query [B, D, H, W] f32  ->  qin [BQ, D] bf16   (LDS tile transpose)
//   grid: (B*H, W/16, D/16), block 256
// ---------------------------------------------------------------------------
__global__ __launch_bounds__(256) void transpose_query(const float* __restrict__ bev,
                                                       bhalf* __restrict__ qin) {
  __shared__ float t[16][17];
  const int tx = threadIdx.x & 15;
  const int ty = threadIdx.x >> 4;
  const long bh = blockIdx.x;              // b*128 + h
  const int  w0 = blockIdx.y * 16;
  const int  d0 = blockIdx.z * 16;
  const long b  = bh >> 7;
  const int  h  = (int)(bh & 127);

  t[ty][tx] = bev[b * (long)(DD * HH * WW) + (long)(d0 + ty) * (HH * WW) + h * WW + w0 + tx];
  __syncthreads();
  // qin[(bh*128 + w0+ty)][d0+tx]  -- contiguous in d across tx (coalesced)
  qin[(bh * 128 + w0 + ty) * (long)DD + d0 + tx] = (bhalf)t[tx][ty];
}

// ---------------------------------------------------------------------------
// Async memory->LDS helpers (CDNA5 ASYNCcnt path). Each lane copies 16 bytes
// straight into LDS without touching VGPR data registers.
// ---------------------------------------------------------------------------
__device__ __forceinline__ void async_copy_b128(const bhalf* gsrc, bhalf* lds_dst) {
  const unsigned long long gaddr = (unsigned long long)gsrc;
  const unsigned ldsoff = (unsigned)(unsigned long long)lds_dst;  // low 32 bits = LDS offset
  asm volatile("global_load_async_to_lds_b128 %0, %1, off"
               :
               : "v"(ldsoff), "v"(gaddr)
               : "memory");
}
__device__ __forceinline__ void wait_async0() {
  asm volatile("s_wait_asynccnt 0x0" ::: "memory");
}

// ---------------------------------------------------------------------------
// Generic bf16 WMMA GEMM:  out[M,N] = epi( A[M,K] * W[N,K]^T + bias (+ add) )
//   A: bf16 (AF32=0, double-buffered async-to-LDS staging) or
//      f32  (AF32=1, VGPR staging with f32->bf16 convert)
//   W: bf16 row-major [N,K]  (lane n holds 16 contiguous K -> B-frag layout)
//   EPI 0: bf16 out, optional full-matrix f32 add (pos embeds)
//   EPI 2: GELU(exact erf) -> bf16 out
//   EPI 3: f32 out scattered into [B, D, H, W] (row index = bq, col = d)
//   Block: 256 thr = 8 waves laid out 2(M) x 4(N). WG tile 64 x 128.
//   Per wave: 2x2 tiles of v_wmma_f32_16x16x32_bf16.
// ---------------------------------------------------------------------------
template <int EPI, int AF32>
__global__ __launch_bounds__(256) void gemm_bf16(const void* __restrict__ Aptr,
                                                 const bhalf* __restrict__ Wt,
                                                 const float* __restrict__ bias,
                                                 const float* __restrict__ addmat,
                                                 void* __restrict__ outp,
                                                 int N, int K) {
  const int tid = threadIdx.x;
  const int L = tid & 31;
  const int wid = tid >> 5;
  const int waverow = wid >> 2;       // 0..1
  const int wavecol = wid & 3;        // 0..3
  const long m0 = (long)blockIdx.x * 64;
  const int n0 = blockIdx.y * 128;

  // A staging tiles: 64 rows x 32 halves, padded row stride 40 halves (80 B:
  // 16B-aligned; 20*L mod 64 enumerates all multiples of 4 -> conflict-free
  // 16-lane ds_load_b128). Two buffers for the async double-buffer pipeline.
  __shared__ alignas(16) bhalf As[2][64 * 40];

  const int arow = L & 15;            // m / n within a 16x16 tile
  const int asel = L >> 4;            // lane-half selector per ISA A/B layouts

  // Per-thread staging coordinates: 256 threads x 16 B == one 64x32 bf16 tile.
  const int sr = tid >> 2;            // staged row 0..63
  const int skp = (tid & 3) * 8;      // staged k-offset in halves {0,8,16,24}

  v8f acc[2][2];
  const v8f zf = {0.f, 0.f, 0.f, 0.f, 0.f, 0.f, 0.f, 0.f};
#pragma unroll
  for (int mt = 0; mt < 2; ++mt)
#pragma unroll
    for (int nt = 0; nt < 2; ++nt) acc[mt][nt] = zf;

  if (AF32 == 0) {
    // Prime the pipeline: stage k-chunk 0 into buffer 0.
    const bhalf* Ab = (const bhalf*)Aptr;
    async_copy_b128(Ab + (m0 + sr) * (long)K + skp, &As[0][sr * 40 + skp]);
  }

  int buf = 0;
  for (int k0 = 0; k0 < K; k0 += 32) {
    if (AF32 == 0) {
      wait_async0();        // own async stores for As[buf] complete
      __syncthreads();      // ...and everyone else's: tile published
      if (k0 + 32 < K) {    // prefetch next chunk into the other buffer
        const bhalf* Ab = (const bhalf*)Aptr;
        async_copy_b128(Ab + (m0 + sr) * (long)K + (k0 + 32) + skp,
                        &As[buf ^ 1][sr * 40 + skp]);
      }
    } else {
      __syncthreads();      // previous reads done before overwrite
      const float* Af = (const float*)Aptr + (m0 + sr) * (long)K + k0 + skp;
      v8f f = *(const v8f*)Af;
      v8bf o;
#pragma unroll
      for (int i = 0; i < 8; ++i) o[i] = (bhalf)f[i];
      *(v8bf*)&As[0][sr * 40 + skp] = o;
      __syncthreads();
    }

    // B fragments (32x16, K x N): lane n = 16 contiguous K values of weight row n
    v16bf bf2[2];
#pragma unroll
    for (int nt = 0; nt < 2; ++nt) {
      const int n = n0 + wavecol * 32 + nt * 16 + arow;
      bf2[nt] = *(const v16bf*)(Wt + (long)n * K + k0 + asel * 16);
    }

    // A fragments (16x32) from LDS: halves {0..15}=K(asel*16..+15)
    v16bf af2[2];
#pragma unroll
    for (int mt = 0; mt < 2; ++mt) {
      const int row = waverow * 32 + mt * 16 + arow;
      const bhalf* ap = &As[buf][row * 40 + asel * 16];
      v8bf lo = *(const v8bf*)ap;
      v8bf hi = *(const v8bf*)(ap + 8);
      af2[mt] = __builtin_shufflevector(lo, hi, 0, 1, 2, 3, 4, 5, 6, 7,
                                        8, 9, 10, 11, 12, 13, 14, 15);
    }

#pragma unroll
    for (int mt = 0; mt < 2; ++mt)
#pragma unroll
      for (int nt = 0; nt < 2; ++nt)
        acc[mt][nt] = __builtin_amdgcn_wmma_f32_16x16x32_bf16(
            false, af2[mt], false, bf2[nt], (short)0, acc[mt][nt], false, false);

    if (AF32 == 0) buf ^= 1;
  }

  // Epilogue. C layout: vgpr r -> M = r + 8*asel, N = lane&15.
#pragma unroll
  for (int mt = 0; mt < 2; ++mt)
#pragma unroll
    for (int nt = 0; nt < 2; ++nt) {
      const long tm0 = m0 + waverow * 32 + mt * 16;
      const int n = n0 + wavecol * 32 + nt * 16 + arow;
      const float bn = bias[n];
#pragma unroll
      for (int r = 0; r < 8; ++r) {
        const long m = tm0 + r + asel * 8;
        float v = acc[mt][nt][r] + bn;
        if (EPI == 0) {
          if (addmat) v += addmat[m * (long)N + n];
          ((bhalf*)outp)[m * (long)N + n] = (bhalf)v;
        } else if (EPI == 2) {
          v = 0.5f * v * (1.0f + erff(v * 0.70710678118654752f));
          ((bhalf*)outp)[m * (long)N + n] = (bhalf)v;
        } else {  // EPI == 3: scatter to [B, D, H, W]
          const long b = m >> 14;          // bq / (H*W)
          const long hw = m & 16383;
          ((float*)outp)[b * (long)(DD * HH * WW) + (long)n * (HH * WW) + hw] = v;
        }
      }
    }
}

// ---------------------------------------------------------------------------
// Attention: per bq -> per-head dot (shfl_xor reduce over 8-lane head groups),
// softmax over P=8, weight raw sampled_feat, write bf16 [BQ, P*F] MLP input.
// One wave per bq; block 256 = 8 bq.
// ---------------------------------------------------------------------------
__global__ __launch_bounds__(256) void attn_kernel(const bhalf* __restrict__ q,
                                                   const bhalf* __restrict__ k,
                                                   const float* __restrict__ sf,
                                                   bhalf* __restrict__ x) {
  const long bq = ((long)blockIdx.x * 256 + threadIdx.x) >> 5;
  const int L = threadIdx.x & 31;
  const int c0 = L * 4;  // 4 channels per lane, all within one head (head = L>>3)

  const v4bf qv = *(const v4bf*)(q + bq * FF + c0);
  float qf[4];
#pragma unroll
  for (int i = 0; i < 4; ++i) qf[i] = (float)qv[i];

  float d8[PP];
#pragma unroll
  for (int p = 0; p < PP; ++p) {
    const v4bf kv = *(const v4bf*)(k + (bq * PP + p) * FF + c0);
    float s = qf[0] * (float)kv[0] + qf[1] * (float)kv[1] +
              qf[2] * (float)kv[2] + qf[3] * (float)kv[3];
    s += __shfl_xor(s, 1, 32);
    s += __shfl_xor(s, 2, 32);
    s += __shfl_xor(s, 4, 32);            // sum across the 8 lanes of this head
    d8[p] = s * 0.17677669529663687f;     // DH^-0.5 = 1/sqrt(32)
  }

  float mx = d8[0];
#pragma unroll
  for (int p = 1; p < PP; ++p) mx = fmaxf(mx, d8[p]);
  float ex[PP], sum = 0.f;
#pragma unroll
  for (int p = 0; p < PP; ++p) { ex[p] = expf(d8[p] - mx); sum += ex[p]; }
  const float inv = 1.0f / sum;

#pragma unroll
  for (int p = 0; p < PP; ++p) {
    const float a = ex[p] * inv;
    const float4 s4 = *(const float4*)(sf + bq * (PP * FF) + p * FF + c0);
    v4bf o;
    o[0] = (bhalf)(a * s4.x);
    o[1] = (bhalf)(a * s4.y);
    o[2] = (bhalf)(a * s4.z);
    o[3] = (bhalf)(a * s4.w);
    *(v4bf*)(x + bq * (PP * FF) + p * FF + c0) = o;
  }
}

// ---------------------------------------------------------------------------
// Launch
// ---------------------------------------------------------------------------
extern "C" void kernel_launch(void* const* d_in, const int* in_sizes, int n_in,
                              void* d_out, int out_size, void* d_ws, size_t ws_size,
                              hipStream_t stream) {
  (void)in_sizes; (void)n_in; (void)out_size; (void)ws_size;
  const float* bev_query   = (const float*)d_in[0];
  const float* sampled     = (const float*)d_in[1];
  const float* bev_pos     = (const float*)d_in[2];   // [BQ, F] contiguous
  const float* pos2d       = (const float*)d_in[3];   // [BQ*P, F] contiguous
  const float* Wq1 = (const float*)d_in[4];  const float* bq1 = (const float*)d_in[5];
  const float* Wq2 = (const float*)d_in[6];  const float* bq2 = (const float*)d_in[7];
  const float* Wk1 = (const float*)d_in[8];  const float* bk1 = (const float*)d_in[9];
  const float* Wk2 = (const float*)d_in[10]; const float* bk2 = (const float*)d_in[11];
  const float* Wc1 = (const float*)d_in[12]; const float* bc1 = (const float*)d_in[13];
  const float* Wc2 = (const float*)d_in[14]; const float* bc2 = (const float*)d_in[15];
  const float* Wc3 = (const float*)d_in[16]; const float* bc3 = (const float*)d_in[17];
  float* out = (float*)d_out;
  char* w = (char*)d_ws;

  // ---- workspace layout (bytes) ----
  const size_t O_Wq1 = 0;
  const size_t O_Wq2 = O_Wq1 + 16384 * 2;
  const size_t O_Wk1 = O_Wq2 + 16384 * 2;
  const size_t O_Wk2 = O_Wk1 + 16384 * 2;
  const size_t O_Wc1 = O_Wk2 + 16384 * 2;
  const size_t O_Wc2 = O_Wc1 + 524288ull * 2;
  const size_t O_Wc3 = O_Wc2 + 262144ull * 2;
  const size_t O_QBF = 2ull << 20;                          // q   bf16 [BQ,F]      16 MB
  const size_t O_KBF = O_QBF + (size_t)BQ * FF * 2;         // k   bf16 [BQ*P,F]   128 MB
  const size_t O_X   = O_KBF + (size_t)BQ * PP * FF * 2;    // tmpk / x bf16       128 MB
  const size_t O_R4  = O_X   + (size_t)BQ * PP * FF * 2;    // qin+tmpq / h1+h2    128 MB
  const size_t O_QIN  = O_R4;
  const size_t O_TMPQ = O_R4 + (size_t)BQ * DD * 2;
  const size_t O_H1   = O_R4;
  const size_t O_H2   = O_R4 + (size_t)BQ * HID * 2;

  bhalf* Wq1b = (bhalf*)(w + O_Wq1); bhalf* Wq2b = (bhalf*)(w + O_Wq2);
  bhalf* Wk1b = (bhalf*)(w + O_Wk1); bhalf* Wk2b = (bhalf*)(w + O_Wk2);
  bhalf* Wc1b = (bhalf*)(w + O_Wc1); bhalf* Wc2b = (bhalf*)(w + O_Wc2);
  bhalf* Wc3b = (bhalf*)(w + O_Wc3);
  bhalf* qbf  = (bhalf*)(w + O_QBF);
  bhalf* kbf  = (bhalf*)(w + O_KBF);
  bhalf* tmpk = (bhalf*)(w + O_X);   bhalf* xrow = (bhalf*)(w + O_X);
  bhalf* qin  = (bhalf*)(w + O_QIN); bhalf* tmpq = (bhalf*)(w + O_TMPQ);
  bhalf* h1   = (bhalf*)(w + O_H1);  bhalf* h2   = (bhalf*)(w + O_H2);

  // 1) weights -> bf16
  cvt_f32_bf16<<<64,   256, 0, stream>>>(Wq1, Wq1b, 16384);
  cvt_f32_bf16<<<64,   256, 0, stream>>>(Wq2, Wq2b, 16384);
  cvt_f32_bf16<<<64,   256, 0, stream>>>(Wk1, Wk1b, 16384);
  cvt_f32_bf16<<<64,   256, 0, stream>>>(Wk2, Wk2b, 16384);
  cvt_f32_bf16<<<2048, 256, 0, stream>>>(Wc1, Wc1b, 524288);
  cvt_f32_bf16<<<1024, 256, 0, stream>>>(Wc2, Wc2b, 262144);
  cvt_f32_bf16<<<256,  256, 0, stream>>>(Wc3, Wc3b, 65536);

  // 2) transpose bev_query -> qin bf16 [BQ, D]
  transpose_query<<<dim3(BB * HH, WW / 16, DD / 16), 256, 0, stream>>>(bev_query, qin);

  // 3) q projections: tmpq = qin*Wq1^T + bq1 + pos_q ; qbf = tmpq*Wq2^T + bq2
  gemm_bf16<0, 0><<<dim3(BQ / 64, 1), 256, 0, stream>>>(qin,  Wq1b, bq1, bev_pos, tmpq, FF, DD);
  gemm_bf16<0, 0><<<dim3(BQ / 64, 1), 256, 0, stream>>>(tmpq, Wq2b, bq2, nullptr, qbf,  FF, FF);

  // 4) k projections: tmpk = sf*Wk1^T + bk1 + pos_k ; kbf = tmpk*Wk2^T + bk2
  gemm_bf16<0, 1><<<dim3(BQ * PP / 64, 1), 256, 0, stream>>>(sampled, Wk1b, bk1, pos2d,  tmpk, FF, FF);
  gemm_bf16<0, 0><<<dim3(BQ * PP / 64, 1), 256, 0, stream>>>(tmpk,    Wk2b, bk2, nullptr, kbf, FF, FF);

  // 5) attention -> x bf16 [BQ, P*F]   (x reuses tmpk's region; tmpk dead)
  attn_kernel<<<BQ / 8, 256, 0, stream>>>(qbf, kbf, sampled, xrow);

  // 6) MLP: h1 = gelu(x*Wc1^T+bc1); h2 = gelu(h1*Wc2^T+bc2); out = h2*Wc3^T+bc3 (transposed)
  gemm_bf16<2, 0><<<dim3(BQ / 64, HID / 128), 256, 0, stream>>>(xrow, Wc1b, bc1, nullptr, h1, HID, PP * FF);
  gemm_bf16<2, 0><<<dim3(BQ / 64, HID / 128), 256, 0, stream>>>(h1,   Wc2b, bc2, nullptr, h2, HID, HID);
  gemm_bf16<3, 0><<<dim3(BQ / 64, 1),         256, 0, stream>>>(h2,   Wc3b, bc3, nullptr, out, DD, HID);
}
